// TransformerLayer_21534966022224
// MI455X (gfx1250) — matrix operand order repaired and checked by reading, not compiled, not run on hardware
//
#include <hip/hip_runtime.h>
#include <hip/hip_bf16.h>

#ifndef USE_ASYNC_LDS
#define USE_ASYNC_LDS 1
#endif

// ---------------------------------------------------------------------------
// Types
// ---------------------------------------------------------------------------
typedef unsigned short bf16_t; // raw bf16 storage
typedef __attribute__((ext_vector_type(8)))  float  v8f;
typedef __attribute__((ext_vector_type(16))) __bf16 v16bf;

union FragU {
    uint4 q[2];
    v16bf v;
};

__device__ __forceinline__ bf16_t f32_to_bf16(float f) {
    unsigned int u = __float_as_uint(f);
    unsigned int r = (u + 0x7FFFu + ((u >> 16) & 1u)) >> 16;
    return (bf16_t)r;
}

// A-fragment (16x32, 16-bit): lane<16 holds M=lane, K = {half*8+0..7, 16+half*8+0..7}
__device__ __forceinline__ v16bf load_frag_a(const bf16_t* base, int row_stride, int lane) {
    int half = (lane >> 4) & 1;
    int m    = lane & 15;
    const bf16_t* p = base + (size_t)m * row_stride + half * 8;
    FragU u;
    u.q[0] = *(const uint4*)(p);
    u.q[1] = *(const uint4*)(p + 16);
    return u.v;
}

// B-fragment (32x16, 16-bit), source stored as rows of N (W[N][K], contiguous K):
// lane -> N = lane&15, elements 0..15 -> K = {half*8+0..7, 16+half*8+0..7}
__device__ __forceinline__ v16bf load_frag_b(const bf16_t* base, int row_stride, int lane) {
    int half = (lane >> 4) & 1;
    int n    = lane & 15;
    const bf16_t* p = base + (size_t)n * row_stride + half * 8;
    FragU u;
    u.q[0] = *(const uint4*)(p);
    u.q[1] = *(const uint4*)(p + 16);
    return u.v;
}

__device__ __forceinline__ v8f wmma_bf16(v16bf a, v16bf b, v8f c) {
    return __builtin_amdgcn_wmma_f32_16x16x32_bf16(false, a, false, b, (short)0, c, false, false);
}

// CDNA5 async global->LDS copy (16B per lane), tracked with ASYNCcnt.
__device__ __forceinline__ void async_load_b128(unsigned lds_off, const bf16_t* gptr) {
#if USE_ASYNC_LDS
    unsigned long long ga = (unsigned long long)(uintptr_t)gptr;
    asm volatile("global_load_async_to_lds_b128 %0, %1, off"
                 :: "v"(lds_off), "v"(ga) : "memory");
#else
    *(uint4*)(uintptr_t)0 = *(const uint4*)gptr; // unused fallback
#endif
}

__device__ __forceinline__ void wait_async_then_barrier() {
#if USE_ASYNC_LDS
    asm volatile("s_wait_asynccnt 0x0" ::: "memory");
#endif
    __syncthreads();
}

// ---------------------------------------------------------------------------
// f32 -> bf16 convert
// ---------------------------------------------------------------------------
__global__ void cvt_f32_bf16_kernel(const float* __restrict__ in, bf16_t* __restrict__ out) {
    size_t i = (size_t)blockIdx.x * blockDim.x + threadIdx.x;
    out[i] = f32_to_bf16(in[i]);
}

// ---------------------------------------------------------------------------
// GEMM: C[M,N] = A[M,K](bf16) * W[N,K]^T(bf16) + bias [+resid][relu]
// block tile 128x128, 8 waves as 2(M)x4(N), wave tile 64x32, K-step 64,
// double-buffered LDS staged with async global->LDS copies.
// ---------------------------------------------------------------------------
#define LDS_STR 72   // 64 + 8 pad (bf16 elems); 144B rows, 16B aligned

__device__ __forceinline__ void gemm_stage(
    bf16_t* AsBuf, bf16_t* BsBuf,
    const bf16_t* __restrict__ A, int lda,
    const bf16_t* __restrict__ Bw, int ldb,
    int row0, int col0, int kt, int tid)
{
#pragma unroll
    for (int j = 0; j < 4; ++j) {
        int i = j * 256 + tid;            // chunk id: 128 rows x 8 chunks
        int r = i >> 3, c = (i & 7) * 8;
        const bf16_t* ga = A  + (size_t)(row0 + r) * lda + kt + c;
        const bf16_t* gb = Bw + (size_t)(col0 + r) * ldb + kt + c;
#if USE_ASYNC_LDS
        async_load_b128((unsigned)(uintptr_t)&AsBuf[r * LDS_STR + c], ga);
        async_load_b128((unsigned)(uintptr_t)&BsBuf[r * LDS_STR + c], gb);
#else
        *(uint4*)&AsBuf[r * LDS_STR + c] = *(const uint4*)ga;
        *(uint4*)&BsBuf[r * LDS_STR + c] = *(const uint4*)gb;
#endif
    }
}

template <bool RESID, bool RELU, bool OUTBF>
__global__ __launch_bounds__(256) void gemm_bf16_kernel(
    const bf16_t* __restrict__ A, int lda,
    const bf16_t* __restrict__ Bw, int ldb,
    const float* __restrict__ bias,
    const float* __restrict__ resid,
    float* __restrict__ Cf, bf16_t* __restrict__ Cb,
    int M, int N, int K)
{
    __shared__ bf16_t As[2][128 * LDS_STR];
    __shared__ bf16_t Bs[2][128 * LDS_STR];

    const int tid  = threadIdx.x;
    const int lane = tid & 31;
    const int wave = tid >> 5;
    const int wm   = wave >> 2;  // 0..1  (64 rows each)
    const int wn   = wave & 3;   // 0..3  (32 cols each)
    const int row0 = blockIdx.y * 128;
    const int col0 = blockIdx.x * 128;

    v8f acc[4][2];
#pragma unroll
    for (int i = 0; i < 4; ++i)
#pragma unroll
        for (int j = 0; j < 2; ++j)
            acc[i][j] = 0.0f;

    int cur = 0;
    gemm_stage(As[0], Bs[0], A, lda, Bw, ldb, row0, col0, 0, tid);

    for (int kt = 0; kt < K; kt += 64) {
        wait_async_then_barrier();
        if (kt + 64 < K)
            gemm_stage(As[cur ^ 1], Bs[cur ^ 1], A, lda, Bw, ldb, row0, col0, kt + 64, tid);

#pragma unroll
        for (int kk = 0; kk < 2; ++kk) {
            v16bf b0 = load_frag_b(&Bs[cur][(wn * 32) * LDS_STR + kk * 32],      LDS_STR, lane);
            v16bf b1 = load_frag_b(&Bs[cur][(wn * 32 + 16) * LDS_STR + kk * 32], LDS_STR, lane);
#pragma unroll
            for (int mi = 0; mi < 4; ++mi) {
                v16bf a = load_frag_a(&As[cur][(wm * 64 + mi * 16) * LDS_STR + kk * 32],
                                      LDS_STR, lane);
                acc[mi][0] = wmma_bf16(a, b0, acc[mi][0]);
                acc[mi][1] = wmma_bf16(a, b1, acc[mi][1]);
            }
        }
        cur ^= 1;
    }

    // epilogue: C layout: VGPR r -> M = r + (lane>=16?8:0), N = lane&15
    const int half = lane >> 4;
    const int nn   = lane & 15;
#pragma unroll
    for (int mi = 0; mi < 4; ++mi) {
#pragma unroll
        for (int ni = 0; ni < 2; ++ni) {
            int gcol = col0 + wn * 32 + ni * 16 + nn;
            float bv = bias[gcol];
#pragma unroll
            for (int r = 0; r < 8; ++r) {
                int grow = row0 + wm * 64 + mi * 16 + r + half * 8;
                size_t idx = (size_t)grow * N + gcol;
                float v = acc[mi][ni][r] + bv;
                if constexpr (RESID) v += resid[idx];
                if constexpr (RELU)  v = fmaxf(v, 0.0f);
                if constexpr (OUTBF) Cb[idx] = f32_to_bf16(v);
                else                 Cf[idx] = v;
            }
        }
    }
    (void)M;
}

// ---------------------------------------------------------------------------
// Flash attention: qkv bf16 [S*B, 3E] -> attn bf16 [S*B, E]
// grid: (S/128, B*H); block: 256 threads (8 waves, 16 q-rows each)
// ---------------------------------------------------------------------------
#define FS   2048
#define FB   2
#define FD   64
#define PST  72
#define QKVLD 3072

__global__ __launch_bounds__(256) void flash_attn_kernel(
    const bf16_t* __restrict__ qkv,
    bf16_t* __restrict__ attn_out,
    float scale)
{
    __shared__ bf16_t Qs[128 * PST];
    __shared__ bf16_t Ks[64 * PST];
    __shared__ bf16_t VTs[FD * PST];
    __shared__ bf16_t Ps[128 * PST];

    const int tid  = threadIdx.x;
    const int lane = tid & 31;
    const int wave = tid >> 5;
    const int bh   = blockIdx.y;
    const int b    = bh & (FB - 1);
    const int h    = bh >> 1;
    const int q0   = blockIdx.x * 128;
    const int half = lane >> 4;
    const int nn   = lane & 15;

    // load Q block 128x64 (async)
#pragma unroll
    for (int j = 0; j < 4; ++j) {
        int i = j * 256 + tid;
        int r = i >> 3, c = (i & 7) * 8;
        int t = (q0 + r) * FB + b;
        const bf16_t* g = qkv + (size_t)t * QKVLD + h * FD + c;
#if USE_ASYNC_LDS
        async_load_b128((unsigned)(uintptr_t)&Qs[r * PST + c], g);
#else
        *(uint4*)&Qs[r * PST + c] = *(const uint4*)g;
#endif
    }

    v8f o_acc[4];
#pragma unroll
    for (int nt = 0; nt < 4; ++nt) o_acc[nt] = 0.0f;
    float m_r[8], l_r[8];
#pragma unroll
    for (int r = 0; r < 8; ++r) { m_r[r] = -1e30f; l_r[r] = 0.0f; }

    for (int j0 = 0; j0 < FS; j0 += 64) {
        __syncthreads();
        // stage K block 64x64 (async)
#pragma unroll
        for (int j = 0; j < 2; ++j) {
            int i = j * 256 + tid;
            int r = i >> 3, c = (i & 7) * 8;
            int t = (j0 + r) * FB + b;
            const bf16_t* g = qkv + (size_t)t * QKVLD + 1024 + h * FD + c;
#if USE_ASYNC_LDS
            async_load_b128((unsigned)(uintptr_t)&Ks[r * PST + c], g);
#else
            *(uint4*)&Ks[r * PST + c] = *(const uint4*)g;
#endif
        }
        // stage V transposed: VTs[d][t] (scalar, layout change)
        for (int i = tid; i < 64 * FD; i += 256) {
            int r = i >> 6, d = i & 63;
            int t = (j0 + r) * FB + b;
            VTs[d * PST + r] = qkv[(size_t)t * QKVLD + 2048 + h * FD + d];
        }
        wait_async_then_barrier();

        // S = Q_wave (16x64) @ K^T (64x64)
        v8f s_acc[4];
#pragma unroll
        for (int nt = 0; nt < 4; ++nt) {
            v8f s = 0.0f;
#pragma unroll
            for (int kk = 0; kk < 2; ++kk) {
                v16bf a  = load_frag_a(&Qs[(wave * 16) * PST + kk * 32], PST, lane);
                v16bf bb = load_frag_b(&Ks[(nt * 16) * PST + kk * 32],   PST, lane);
                s = wmma_bf16(a, bb, s);
            }
            s_acc[nt] = s;
        }

        // online softmax per owned row (r, half)
#pragma unroll
        for (int r = 0; r < 8; ++r) {
            float s0 = s_acc[0][r] * scale;
            float s1 = s_acc[1][r] * scale;
            float s2 = s_acc[2][r] * scale;
            float s3 = s_acc[3][r] * scale;
            float mx = fmaxf(fmaxf(s0, s1), fmaxf(s2, s3));
            for (int off = 1; off < 16; off <<= 1)
                mx = fmaxf(mx, __shfl_xor(mx, off, 32));
            float m_new = fmaxf(m_r[r], mx);
            float p0 = __expf(s0 - m_new);
            float p1 = __expf(s1 - m_new);
            float p2 = __expf(s2 - m_new);
            float p3 = __expf(s3 - m_new);
            float rs = p0 + p1 + p2 + p3;
            for (int off = 1; off < 16; off <<= 1)
                rs += __shfl_xor(rs, off, 32);
            float corr = __expf(m_r[r] - m_new);
            l_r[r] = l_r[r] * corr + rs;
            m_r[r] = m_new;
#pragma unroll
            for (int nt = 0; nt < 4; ++nt)
                o_acc[nt][r] = o_acc[nt][r] * corr;
            bf16_t* pr = &Ps[(wave * 16 + r + half * 8) * PST];
            pr[0 * 16 + nn] = f32_to_bf16(p0);
            pr[1 * 16 + nn] = f32_to_bf16(p1);
            pr[2 * 16 + nn] = f32_to_bf16(p2);
            pr[3 * 16 + nn] = f32_to_bf16(p3);
        }
        __syncthreads();

        // O += P (16x64) @ V (64x64)
#pragma unroll
        for (int nt = 0; nt < 4; ++nt) {
            v8f o = o_acc[nt];
#pragma unroll
            for (int kk = 0; kk < 2; ++kk) {
                v16bf a  = load_frag_a(&Ps[(wave * 16) * PST + kk * 32], PST, lane);
                v16bf bb = load_frag_b(&VTs[(nt * 16) * PST + kk * 32], PST, lane);
                o = wmma_bf16(a, bb, o);
            }
            o_acc[nt] = o;
        }
    }

    // final normalize + store
#pragma unroll
    for (int r = 0; r < 8; ++r) {
        float inv_l = 1.0f / l_r[r];
        int s = q0 + wave * 16 + r + half * 8;
        int t = s * FB + b;
#pragma unroll
        for (int nt = 0; nt < 4; ++nt) {
            int col = h * FD + nt * 16 + nn;
            attn_out[(size_t)t * 1024 + col] = f32_to_bf16(o_acc[nt][r] * inv_l);
        }
    }
}

// ---------------------------------------------------------------------------
// LayerNorm: one block per row (1024 cols)
// ---------------------------------------------------------------------------
__global__ __launch_bounds__(256) void layernorm_kernel(
    const float* __restrict__ X,
    const float* __restrict__ w,
    const float* __restrict__ bp,
    float* __restrict__ Yf,
    bf16_t* __restrict__ Yb,
    int ncols)
{
    __shared__ float red[256];
    const int row = blockIdx.x;
    const int tid = threadIdx.x;
    const float* xr = X + (size_t)row * ncols;

    float s = 0.0f;
    for (int c = tid; c < ncols; c += 256) s += xr[c];
    red[tid] = s; __syncthreads();
    for (int st = 128; st > 0; st >>= 1) {
        if (tid < st) red[tid] += red[tid + st];
        __syncthreads();
    }
    float mean = red[0] / ncols;
    __syncthreads();

    float v = 0.0f;
    for (int c = tid; c < ncols; c += 256) { float d = xr[c] - mean; v += d * d; }
    red[tid] = v; __syncthreads();
    for (int st = 128; st > 0; st >>= 1) {
        if (tid < st) red[tid] += red[tid + st];
        __syncthreads();
    }
    float var = red[0] / ncols;
    float inv = 1.0f / sqrtf(var + 1e-12f);

    for (int c = tid; c < ncols; c += 256) {
        float y = w[c] * ((xr[c] - mean) * inv) + bp[c];
        size_t idx = (size_t)row * ncols + c;
        if (Yf) Yf[idx] = y;
        if (Yb) Yb[idx] = f32_to_bf16(y);
    }
}

// ---------------------------------------------------------------------------
// launch
// ---------------------------------------------------------------------------
extern "C" void kernel_launch(void* const* d_in, const int* in_sizes, int n_in,
                              void* d_out, int out_size, void* d_ws, size_t ws_size,
                              hipStream_t stream) {
    (void)in_sizes; (void)n_in; (void)out_size; (void)ws_size;
    const int S = 2048, B = 2, E = 1024, FF = 4096;
    const int NT = S * B;           // 4096 tokens

    const float* x         = (const float*)d_in[0];
    const float* in_proj_w = (const float*)d_in[1];
    const float* in_proj_b = (const float*)d_in[2];
    const float* out_w     = (const float*)d_in[3];
    const float* out_b     = (const float*)d_in[4];
    const float* fc1_w     = (const float*)d_in[5];
    const float* fc1_b     = (const float*)d_in[6];
    const float* fc2_w     = (const float*)d_in[7];
    const float* fc2_b     = (const float*)d_in[8];
    const float* ln1_w     = (const float*)d_in[9];
    const float* ln1_b     = (const float*)d_in[10];
    const float* ln2_w     = (const float*)d_in[11];
    const float* ln2_b     = (const float*)d_in[12];
    float* out = (float*)d_out;

    char* wsp = (char*)d_ws;
    size_t off = 0;
    auto alloc = [&](size_t bytes) -> void* {
        void* p = wsp + off;
        off += (bytes + 255) & ~(size_t)255;
        return p;
    };

    bf16_t* xb    = (bf16_t*)alloc((size_t)NT * E * 2);
    bf16_t* ipwb  = (bf16_t*)alloc((size_t)3 * E * E * 2);
    bf16_t* outwb = (bf16_t*)alloc((size_t)E * E * 2);
    bf16_t* fc1wb = (bf16_t*)alloc((size_t)FF * E * 2);
    bf16_t* fc2wb = (bf16_t*)alloc((size_t)E * FF * 2);
    bf16_t* qkvb  = (bf16_t*)alloc((size_t)NT * 3 * E * 2);
    bf16_t* attnb = (bf16_t*)alloc((size_t)NT * E * 2);
    float*  y1    = (float*)alloc((size_t)NT * E * 4);
    bf16_t* hb    = (bf16_t*)alloc((size_t)NT * E * 2);
    float*  hf    = (float*)alloc((size_t)NT * E * 4);
    bf16_t* ff1b  = (bf16_t*)alloc((size_t)NT * FF * 2);
    float*  y2    = (float*)alloc((size_t)NT * E * 4);

    // 1) bf16 conversions
    cvt_f32_bf16_kernel<<<(NT * E) / 256, 256, 0, stream>>>(x, xb);
    cvt_f32_bf16_kernel<<<(3 * E * E) / 256, 256, 0, stream>>>(in_proj_w, ipwb);
    cvt_f32_bf16_kernel<<<(E * E) / 256, 256, 0, stream>>>(out_w, outwb);
    cvt_f32_bf16_kernel<<<(FF * E) / 256, 256, 0, stream>>>(fc1_w, fc1wb);
    cvt_f32_bf16_kernel<<<(E * FF) / 256, 256, 0, stream>>>(fc2_w, fc2wb);

    // 2) qkv = x @ in_proj_w^T + b  -> bf16 [NT, 3E]
    gemm_bf16_kernel<false, false, true><<<dim3((3 * E) / 128, NT / 128), 256, 0, stream>>>(
        xb, E, ipwb, E, in_proj_b, nullptr, nullptr, qkvb, NT, 3 * E, E);

    // 3) flash attention -> attnb bf16 [NT, E]
    flash_attn_kernel<<<dim3(S / 128, B * 16), 256, 0, stream>>>(qkvb, attnb, 0.125f);

    // 4) y1 = attn @ out_w^T + out_b + x
    gemm_bf16_kernel<true, false, false><<<dim3(E / 128, NT / 128), 256, 0, stream>>>(
        attnb, E, outwb, E, out_b, x, y1, nullptr, NT, E, E);

    // 5) h = LN1(y1)  (f32 + bf16)
    layernorm_kernel<<<NT, 256, 0, stream>>>(y1, ln1_w, ln1_b, hf, hb, E);

    // 6) ff1 = relu(h @ fc1_w^T + fc1_b)  -> bf16 [NT, FF]
    gemm_bf16_kernel<false, true, true><<<dim3(FF / 128, NT / 128), 256, 0, stream>>>(
        hb, E, fc1wb, E, fc1_b, nullptr, nullptr, ff1b, NT, FF, E);

    // 7) y2 = ff1 @ fc2_w^T + fc2_b + h
    gemm_bf16_kernel<true, false, false><<<dim3(E / 128, NT / 128), 256, 0, stream>>>(
        ff1b, FF, fc2wb, FF, fc2_b, hf, y2, nullptr, NT, E, FF);

    // 8) out = LN2(y2)
    layernorm_kernel<<<NT, 256, 0, stream>>>(y2, ln2_w, ln2_b, out, nullptr, E);
}
